// VectorQuantizer_10170482556965
// MI455X (gfx1250) — matrix operand order repaired and checked by reading, not compiled, not run on hardware
//
#include <hip/hip_runtime.h>
#include <math.h>

typedef __attribute__((ext_vector_type(16))) __bf16 v16bf;
typedef __attribute__((ext_vector_type(8)))  float  v8f;
typedef __attribute__((ext_vector_type(4)))  float  f4;
typedef __attribute__((ext_vector_type(4)))  unsigned int u4;

#define NUM_EMB 1024
#define EMB_DIM 64
#define ROWS_PER_WG 128
#define CHUNK 256          // codebook entries staged in LDS per pass (32 KB bf16)

// ---------------- prep: bf16 codebook + ||e||^2, zero scratch ----------------
__global__ __launch_bounds__(256)
void vq_prep(const float* __restrict__ w, unsigned short* __restrict__ wbf,
             float* __restrict__ e2, int* __restrict__ counts,
             float* __restrict__ sse) {
  int k = blockIdx.x * 256 + threadIdx.x;
  if (k < NUM_EMB) {
    const float* wr = w + (size_t)k * EMB_DIM;
    float s = 0.f;
    #pragma unroll
    for (int j = 0; j < EMB_DIM; ++j) {
      float v = wr[j];
      s += v * v;
      __bf16 bv = (__bf16)v;
      wbf[(size_t)k * EMB_DIM + j] = __builtin_bit_cast(unsigned short, bv);
    }
    e2[k] = s;
    counts[k] = 0;
  }
  if (k == 0) *sse = 0.f;
}

// ---------------- main: WMMA distance + argmin + all big outputs -------------
__global__ __launch_bounds__(256)
void vq_main(const float* __restrict__ x, const float* __restrict__ w,
             const unsigned short* __restrict__ wbf, const float* __restrict__ e2,
             float* __restrict__ outQ, float* __restrict__ outEnc,
             int* __restrict__ counts, float* __restrict__ sseAcc) {
  __shared__ __align__(32) unsigned short Wlds[CHUNK * EMB_DIM]; // 32 KB
  __shared__ int   idxLds[ROWS_PER_WG];
  __shared__ float redLds[8];

  const int t    = threadIdx.x;
  const int lane = t & 31;
  const int wave = t >> 5;
  const int half = lane >> 4;   // which 16-lane half
  const int col  = lane & 15;

  // ---- A fragments: this wave owns 16 rows; lane loads row (base+col) ----
  // A 16x32 layout (contiguous, matches B): element s <-> K = 16h+s
  const int rowBase = blockIdx.x * ROWS_PER_WG + wave * 16;
  v16bf a0, a1;
  {
    const float* xr = x + (size_t)(rowBase + col) * EMB_DIM + 16 * half;
    #pragma unroll
    for (int g = 0; g < 4; ++g) {
      f4 lo = *reinterpret_cast<const f4*>(xr + (g >> 1) * 32 + (g & 1) * 8);
      f4 hi = *reinterpret_cast<const f4*>(xr + (g >> 1) * 32 + (g & 1) * 8 + 4);
      #pragma unroll
      for (int j = 0; j < 4; ++j) {
        if (g < 2) { a0[g * 8 + j] = (__bf16)lo[j]; a0[g * 8 + 4 + j] = (__bf16)hi[j]; }
        else       { a1[(g - 2) * 8 + j] = (__bf16)lo[j]; a1[(g - 2) * 8 + 4 + j] = (__bf16)hi[j]; }
      }
    }
  }

  float best[8];
  int   bidx[8];
  #pragma unroll
  for (int r = 0; r < 8; ++r) { best[r] = 3.4e38f; bidx[r] = 0; }

  for (int chunk = 0; chunk < NUM_EMB / CHUNK; ++chunk) {
    __syncthreads();
    { // stage 32 KB of bf16 codebook: 2048 x 16B, 8 per thread
      const u4* src = reinterpret_cast<const u4*>(wbf + (size_t)chunk * CHUNK * EMB_DIM);
      u4* dst = reinterpret_cast<u4*>(Wlds);
      #pragma unroll
      for (int i = 0; i < 8; ++i) dst[t + i * 256] = src[t + i * 256];
    }
    __syncthreads();

    #pragma unroll 4
    for (int nt = 0; nt < CHUNK / 16; ++nt) {
      const int nl = nt * 16 + col;          // local codebook row (B column)
      const int ng = chunk * CHUNK + nl;     // global codebook index
      // B 32x16 bf16 layout: lanes 0-15 K=0..15, lanes 16-31 K=16..31 (per K-step)
      const v16bf b0 = *reinterpret_cast<const v16bf*>(&Wlds[nl * EMB_DIM + 16 * half]);
      const v16bf b1 = *reinterpret_cast<const v16bf*>(&Wlds[nl * EMB_DIM + 32 + 16 * half]);
      v8f c = {};
      c = __builtin_amdgcn_wmma_f32_16x16x32_bf16(false, a0, false, b0, (short)0, c, false, false);
      c = __builtin_amdgcn_wmma_f32_16x16x32_bf16(false, a1, false, b1, (short)0, c, false, false);
      const float e2v = e2[ng];
      #pragma unroll
      for (int r = 0; r < 8; ++r) {          // score = ||e||^2 - 2 x.e
        float s = __builtin_fmaf(-2.f, c[r], e2v);
        if (s < best[r]) { best[r] = s; bidx[r] = ng; }
      }
    }
  }

  // ---- argmin across the 16 lanes of each half (first index wins ties) ----
  #pragma unroll
  for (int r = 0; r < 8; ++r) {
    float bv = best[r]; int bi = bidx[r];
    #pragma unroll
    for (int mask = 1; mask < 16; mask <<= 1) {
      float ov = __shfl_xor(bv, mask, 32);
      int   oi = __shfl_xor(bi, mask, 32);
      if (ov < bv || (ov == bv && oi < bi)) { bv = ov; bi = oi; }
    }
    if (col == 0) idxLds[wave * 16 + half * 8 + r] = bi;  // M = r + 8*half
  }
  __syncthreads();

  const size_t row0 = (size_t)blockIdx.x * ROWS_PER_WG;

  // ---- quantized = weight[idx] (nontemporal) + SSE partial ----
  float sse = 0.f;
  for (int i = t; i < ROWS_PER_WG * EMB_DIM / 4; i += 256) {
    int mr = i >> 4;
    int k4 = (i & 15) << 2;
    int id = idxLds[mr];
    f4 wv = *reinterpret_cast<const f4*>(w + (size_t)id * EMB_DIM + k4);
    f4 xv = *reinterpret_cast<const f4*>(x + (row0 + mr) * EMB_DIM + k4);
    f4 d  = wv - xv;
    sse += d[0] * d[0] + d[1] * d[1] + d[2] * d[2] + d[3] * d[3];
    __builtin_nontemporal_store(wv, reinterpret_cast<f4*>(outQ + (row0 + mr) * EMB_DIM + k4));
  }
  #pragma unroll
  for (int mask = 1; mask < 32; mask <<= 1) sse += __shfl_xor(sse, mask, 32);
  if (lane == 0) redLds[wave] = sse;
  __syncthreads();
  if (t == 0) {
    float s = 0.f;
    #pragma unroll
    for (int i = 0; i < 8; ++i) s += redLds[i];
    atomicAdd(sseAcc, s);
  }

  // ---- histogram ----
  if (t < ROWS_PER_WG) atomicAdd(&counts[idxLds[t]], 1);

  // ---- one-hot encodings: 512 KB per WG, nontemporal streaming stores ----
  for (int i = t; i < ROWS_PER_WG * NUM_EMB / 4; i += 256) {
    int mr = i >> 8;
    int q4 = i & 255;
    int id = idxLds[mr];
    f4 v = {0.f, 0.f, 0.f, 0.f};
    if ((id >> 2) == q4) v[id & 3] = 1.0f;
    __builtin_nontemporal_store(v, reinterpret_cast<f4*>(outEnc + (row0 + mr) * NUM_EMB + (q4 << 2)));
  }
}

// ---------------- final: loss & perplexity scalars ----------------
__global__ __launch_bounds__(256)
void vq_final(const int* __restrict__ counts, const float* __restrict__ sseAcc,
              float* __restrict__ lossOut, float* __restrict__ perpOut, int N) {
  __shared__ float red[256];
  float h = 0.f;
  float invN = 1.0f / (float)N;
  for (int k = threadIdx.x; k < NUM_EMB; k += 256) {
    float p = (float)counts[k] * invN;
    h += p * logf(p + 1e-10f);
  }
  red[threadIdx.x] = h;
  __syncthreads();
  for (int s = 128; s > 0; s >>= 1) {
    if (threadIdx.x < s) red[threadIdx.x] += red[threadIdx.x + s];
    __syncthreads();
  }
  if (threadIdx.x == 0) {
    *perpOut = expf(-red[0]);
    *lossOut = 1.25f * (*sseAcc) / ((float)N * (float)EMB_DIM);
  }
}

// ---------------- launch ----------------
extern "C" void kernel_launch(void* const* d_in, const int* in_sizes, int n_in,
                              void* d_out, int out_size, void* d_ws, size_t ws_size,
                              hipStream_t stream) {
  const float* x = (const float*)d_in[0];
  const float* w = (const float*)d_in[1];
  const int N = in_sizes[0] / EMB_DIM;   // 131072

  float* out    = (float*)d_out;
  float* lossP  = out;                               // [0]
  float* outQ   = out + 1;                           // [1 .. 1+N*64)
  float* perpP  = out + 1 + (size_t)N * EMB_DIM;     // [1+N*64]
  float* outEnc = out + 2 + (size_t)N * EMB_DIM;     // [2+N*64 ..)

  char* ws = (char*)d_ws;
  unsigned short* wbf = (unsigned short*)ws;                 // 128 KB
  float* e2   = (float*)(ws + 131072);                       // 4 KB
  int*   cnts = (int*)  (ws + 131072 + 4096);                // 4 KB
  float* sse  = (float*)(ws + 131072 + 8192);                // 4 B

  vq_prep<<<dim3((NUM_EMB + 255) / 256), dim3(256), 0, stream>>>(w, wbf, e2, cnts, sse);
  vq_main<<<dim3(N / ROWS_PER_WG), dim3(256), 0, stream>>>(x, w, wbf, e2, outQ, outEnc, cnts, sse);
  vq_final<<<dim3(1), dim3(256), 0, stream>>>(cnts, sse, lossP, perpP, N);
}
